// GCN_57354993270871
// MI455X (gfx1250) — matrix operand order repaired and checked by reading, not compiled, not run on hardware
//
#include <hip/hip_runtime.h>
#include <hip/hip_bf16.h>

// ---------------- constants from the reference ----------------
#define FIN   256
#define HH    2
#define C1    200
#define C2    100
#define HC1   (HH*C1)   // 400
#define HC2   (HH*C2)   // 200
#define GG    64
#define SLOPE 0.2f
#define EPS_  1e-5f
#define KP2   416       // 400 padded to multiple of 32
#define NP2   208       // 200 padded to multiple of 16

typedef __attribute__((ext_vector_type(16))) _Float16 v16h;
typedef __attribute__((ext_vector_type(8)))  _Float16 v8h;
typedef __attribute__((ext_vector_type(8)))  float    v8f;

// order-preserving float<->uint encoding for atomicMax on floats
__device__ __forceinline__ unsigned fenc(float f) {
    unsigned b = __float_as_uint(f);
    return (b & 0x80000000u) ? ~b : (b | 0x80000000u);
}
__device__ __forceinline__ float fdec(unsigned u) {
    unsigned b = (u & 0x80000000u) ? (u ^ 0x80000000u) : ~u;
    return __uint_as_float(b);
}

// ---------- f32 -> f16 convert with K padding and optional relu ----------
__global__ void k_cvt_f16(const float* __restrict__ in, _Float16* __restrict__ out,
                          int rows, int Kin, int Kp, int do_relu) {
    long long i = (long long)blockIdx.x * blockDim.x + threadIdx.x;
    long long total = (long long)rows * Kp;
    if (i >= total) return;
    int k = (int)(i % Kp);
    long long r = i / Kp;
    float v = 0.f;
    if (k < Kin) {
        v = in[r * Kin + k];
        if (do_relu) v = v > 0.f ? v : 0.f;
    }
    out[i] = (_Float16)v;
}

// ---------- W[K,Nout] f32 -> Wt[Np,Kp] f16 (transposed, zero padded) ----------
__global__ void k_wT_f16(const float* __restrict__ W, _Float16* __restrict__ Wt,
                         int K, int Nout, int Kp, int Np) {
    long long i = (long long)blockIdx.x * blockDim.x + threadIdx.x;
    long long total = (long long)Np * Kp;
    if (i >= total) return;
    int k = (int)(i % Kp);
    int n = (int)(i / Kp);
    float v = (n < Nout && k < K) ? W[(size_t)k * Nout + n] : 0.f;
    Wt[i] = (_Float16)v;
}

// ---------- WMMA GEMM: C[M,nvalid] = A[M,Kp](f16) x Bt[Np,Kp](f16, N-major) ----------
// one wave (32 threads) per 16x16 output tile; f32 accumulate
__global__ void k_gemm_wmma(const _Float16* __restrict__ A,
                            const _Float16* __restrict__ Bt,
                            float* __restrict__ C,
                            int Kp, int ntiles, int nvalid, int ldc) {
    int tile = blockIdx.x;
    int mt = tile / ntiles, nt = tile % ntiles;
    int lane = threadIdx.x;
    int hi = lane >> 4;          // 0 for lanes 0-15, 1 for lanes 16-31
    int lr = lane & 15;
    const _Float16* arow = A  + (size_t)(mt * 16 + lr) * Kp;
    const _Float16* brow = Bt + (size_t)(nt * 16 + lr) * Kp;
    v8f acc = {};
    for (int kk = 0; kk < Kp; kk += 32) {
        // A 16x32 f16 layout: lanes<16 get K 0..7 then 16..23; lanes>=16 get 8..15 then 24..31
        v8h a0 = *(const v8h*)(arow + kk + hi * 8);
        v8h a1 = *(const v8h*)(arow + kk + 16 + hi * 8);
        // B 32x16 f16 layout: lanes<16 get K 0..7 then 16..23; lanes>=16 get 8..15 then 24..31 (N = lane&15)
        v8h b0 = *(const v8h*)(brow + kk + hi * 8);
        v8h b1 = *(const v8h*)(brow + kk + 16 + hi * 8);
        v16h a, b;
#pragma unroll
        for (int t = 0; t < 8; ++t) { a[t] = a0[t]; a[t + 8] = a1[t]; b[t] = b0[t]; b[t + 8] = b1[t]; }
        acc = __builtin_amdgcn_wmma_f32_16x16x32_f16(false, a, false, b, (short)0, acc, false, false);
    }
    int ncol = nt * 16 + lr;
    if (ncol < nvalid) {
#pragma unroll
        for (int v = 0; v < 8; ++v)
            C[(size_t)(mt * 16 + hi * 8 + v) * ldc + ncol] = acc[v];
    }
}

// ---------- per-node attention logits: als/ald[n,h] = dot(xw[n,h,:], a_*[h,:]) ----------
__global__ void k_attn(const float* __restrict__ xw, const float* __restrict__ a_src,
                       const float* __restrict__ a_dst, float* __restrict__ als,
                       float* __restrict__ ald, int N, int C) {
    int i = blockIdx.x * blockDim.x + threadIdx.x;
    if (i >= N * HH) return;
    int h = i & 1, n = i >> 1;
    const float* row = xw + (size_t)n * (HH * C) + h * C;
    const float* as = a_src + h * C;
    const float* ad = a_dst + h * C;
    float s1 = 0.f, s2 = 0.f;
    for (int c = 0; c < C; ++c) { float v = row[c]; s1 += v * as[c]; s2 += v * ad[c]; }
    als[i] = s1; ald[i] = s2;
}

// ---------- edge pass 1: segment max of leakyrelu(als[src]+ald[dst]) over dst ----------
__global__ void k_edge_max(const float* __restrict__ als, const float* __restrict__ ald,
                           const int* __restrict__ src, const int* __restrict__ dst,
                           unsigned* __restrict__ menc, int E0, int Ep) {
    int i = blockIdx.x * blockDim.x + threadIdx.x;
    if (i >= Ep * HH) return;
    int h = i & 1, e = i >> 1;
    int s, d;
    if (e < E0) { s = src[e]; d = dst[e]; } else { s = d = e - E0; }
    float v = als[s * HH + h] + ald[d * HH + h];
    v = v > 0.f ? v : SLOPE * v;
    atomicMax(&menc[d * HH + h], fenc(v));
}

// ---------- edge pass 2: p = exp(e - m[dst]); z[dst] += p ----------
__global__ void k_edge_p(const float* __restrict__ als, const float* __restrict__ ald,
                         const int* __restrict__ src, const int* __restrict__ dst,
                         const unsigned* __restrict__ menc, float* __restrict__ p,
                         float* __restrict__ z, int E0, int Ep) {
    int i = blockIdx.x * blockDim.x + threadIdx.x;
    if (i >= Ep * HH) return;
    int h = i & 1, e = i >> 1;
    int s, d;
    if (e < E0) { s = src[e]; d = dst[e]; } else { s = d = e - E0; }
    float v = als[s * HH + h] + ald[d * HH + h];
    v = v > 0.f ? v : SLOPE * v;
    float pe = __expf(v - fdec(menc[d * HH + h]));
    p[i] = pe;
    atomicAdd(&z[d * HH + h], pe);
}

// ---------- init out with bias ----------
__global__ void k_init_bias(float* __restrict__ out, const float* __restrict__ b,
                            int N, int C) {
    long long i = (long long)blockIdx.x * blockDim.x + threadIdx.x;
    if (i >= (long long)N * C) return;
    out[i] = b[i % C];
}

// ---------- edge pass 3: out[dst] += (p/z[dst]) * xw[src]  (per channel) ----------
__global__ void k_edge_aggr(const float* __restrict__ p, const float* __restrict__ z,
                            const float* __restrict__ xw, float* __restrict__ out,
                            const int* __restrict__ src, const int* __restrict__ dst,
                            int E0, int Ep, int C) {
    int c4 = (C + 3) >> 2;
    long long total = (long long)Ep * HH * c4;
    long long i = (long long)blockIdx.x * blockDim.x + threadIdx.x;
    if (i >= total) return;
    int t = (int)(i % c4);
    long long eh = i / c4;
    int h = (int)(eh & 1);
    int e = (int)(eh >> 1);
    int s, d;
    if (e < E0) { s = src[e]; d = dst[e]; } else { s = d = e - E0; }
    float alpha = p[(size_t)e * HH + h] / z[d * HH + h];
    int ld = HH * C;
    const float* srow = xw + (size_t)s * ld + h * C;
    float* drow = out + (size_t)d * ld + h * C;
    for (int c = t * 4; c < t * 4 + 4 && c < C; ++c)
        atomicAdd(&drow[c], alpha * srow[c]);
}

// ---------- GraphNorm helpers ----------
__global__ void k_counts(const int* __restrict__ batch, float* __restrict__ counts, int N) {
    int i = blockIdx.x * blockDim.x + threadIdx.x;
    if (i >= N) return;
    atomicAdd(&counts[batch[i]], 1.0f);
}
__global__ void k_gn_sum(const float* __restrict__ h, const int* __restrict__ batch,
                         float* __restrict__ gsum, int N) {
    long long i = (long long)blockIdx.x * blockDim.x + threadIdx.x;
    if (i >= (long long)N * HC2) return;
    int c = (int)(i % HC2); int n = (int)(i / HC2);
    atomicAdd(&gsum[batch[n] * HC2 + c], h[i]);
}
__global__ void k_gn_center(float* __restrict__ h, const int* __restrict__ batch,
                            const float* __restrict__ gsum, const float* __restrict__ counts,
                            const float* __restrict__ ms, float* __restrict__ gvar, int N) {
    long long i = (long long)blockIdx.x * blockDim.x + threadIdx.x;
    if (i >= (long long)N * HC2) return;
    int c = (int)(i % HC2); int n = (int)(i / HC2);
    int g = batch[n];
    float mean = gsum[g * HC2 + c] / counts[g];
    float t = h[i] - mean * ms[c];
    h[i] = t;
    atomicAdd(&gvar[g * HC2 + c], t * t);
}
__global__ void k_gn_final(const float* __restrict__ h, const int* __restrict__ batch,
                           const float* __restrict__ gvar, const float* __restrict__ counts,
                           const float* __restrict__ w, const float* __restrict__ b,
                           float* __restrict__ pooled, int N) {
    long long i = (long long)blockIdx.x * blockDim.x + threadIdx.x;
    if (i >= (long long)N * HC2) return;
    int c = (int)(i % HC2); int n = (int)(i / HC2);
    int g = batch[n];
    float var = gvar[g * HC2 + c] / counts[g];
    float y = h[i] * rsqrtf(var + EPS_);
    y = w[c] * y + b[c];
    y = y > 0.f ? y : 0.f;           // relu
    atomicAdd(&pooled[g * HC2 + c], y);
}
__global__ void k_final(const float* __restrict__ pooled, const float* __restrict__ counts,
                        const float* __restrict__ lin_w, const float* __restrict__ lin_b,
                        float* __restrict__ out) {
    int i = blockIdx.x * blockDim.x + threadIdx.x;
    if (i >= GG * 2) return;
    int j = i & 1, g = i >> 1;
    float inv = 1.0f / counts[g];
    float acc = lin_b[j];
    for (int c = 0; c < HC2; ++c)
        acc += pooled[g * HC2 + c] * inv * lin_w[c * 2 + j];
    out[i] = 1.0f / (1.0f + __expf(-acc));
}

static inline unsigned blks(long long total, int bs) { return (unsigned)((total + bs - 1) / bs); }

extern "C" void kernel_launch(void* const* d_in, const int* in_sizes, int n_in,
                              void* d_out, int out_size, void* d_ws, size_t ws_size,
                              hipStream_t stream) {
    const float* x     = (const float*)d_in[0];
    const int*   ei    = (const int*)d_in[1];
    const int*   batch = (const int*)d_in[2];
    const float* W1    = (const float*)d_in[3];
    const float* a_s1  = (const float*)d_in[4];
    const float* a_d1  = (const float*)d_in[5];
    const float* b1    = (const float*)d_in[6];
    const float* W2    = (const float*)d_in[7];
    const float* a_s2  = (const float*)d_in[8];
    const float* a_d2  = (const float*)d_in[9];
    const float* b2    = (const float*)d_in[10];
    const float* gn_w  = (const float*)d_in[11];
    const float* gn_b  = (const float*)d_in[12];
    const float* gn_ms = (const float*)d_in[13];
    const float* lin_w = (const float*)d_in[14];
    const float* lin_b = (const float*)d_in[15];

    const int N  = in_sizes[0] / FIN;   // 50000 (divisible by 16)
    const int E0 = in_sizes[1] / 2;     // 800000
    const int Ep = E0 + N;              // with self loops
    const int* src = ei;
    const int* dst = ei + E0;

    // ---- workspace carve-out (256B aligned sub-buffers) ----
    char* ws = (char*)d_ws;
    size_t off = 0;
    auto carve = [&](size_t bytes) -> void* {
        void* pp = ws + off;
        off = (off + bytes + 255) & ~(size_t)255;
        return pp;
    };
    _Float16* XH1 = (_Float16*)carve((size_t)N * FIN * 2);
    _Float16* W1T = (_Float16*)carve((size_t)HC1 * FIN * 2);
    float*    XW1 = (float*)   carve((size_t)N * HC1 * 4);
    float*    OUT1= (float*)   carve((size_t)N * HC1 * 4);
    _Float16* XH2 = (_Float16*)carve((size_t)N * KP2 * 2);
    _Float16* W2T = (_Float16*)carve((size_t)NP2 * KP2 * 2);
    float*    XW2 = (float*)   carve((size_t)N * HC2 * 4);
    float*    OUT2= (float*)   carve((size_t)N * HC2 * 4);
    float*    ALS = (float*)   carve((size_t)N * HH * 4);
    float*    ALD = (float*)   carve((size_t)N * HH * 4);
    unsigned* MENC= (unsigned*)carve((size_t)N * HH * 4);
    float*    Z   = (float*)   carve((size_t)N * HH * 4);
    float*    P   = (float*)   carve((size_t)Ep * HH * 4);
    float*    CNT = (float*)   carve(GG * 4);
    float*    GSUM= (float*)   carve(GG * HC2 * 4);
    float*    GVAR= (float*)   carve(GG * HC2 * 4);
    float*    POOL= (float*)   carve(GG * HC2 * 4);

    const int BS = 256;
    const int mtiles = N / 16;

    // ================= layer 1 =================
    k_cvt_f16<<<blks((long long)N * FIN, BS), BS, 0, stream>>>(x, XH1, N, FIN, FIN, 0);
    k_wT_f16 <<<blks((long long)HC1 * FIN, BS), BS, 0, stream>>>(W1, W1T, FIN, HC1, FIN, HC1);
    k_gemm_wmma<<<mtiles * (HC1 / 16), 32, 0, stream>>>(XH1, W1T, XW1, FIN, HC1 / 16, HC1, HC1);
    k_attn<<<blks((long long)N * HH, BS), BS, 0, stream>>>(XW1, a_s1, a_d1, ALS, ALD, N, C1);
    hipMemsetAsync(MENC, 0, (size_t)N * HH * 4, stream);
    hipMemsetAsync(Z,    0, (size_t)N * HH * 4, stream);
    k_edge_max<<<blks((long long)Ep * HH, BS), BS, 0, stream>>>(ALS, ALD, src, dst, MENC, E0, Ep);
    k_edge_p  <<<blks((long long)Ep * HH, BS), BS, 0, stream>>>(ALS, ALD, src, dst, MENC, P, Z, E0, Ep);
    k_init_bias<<<blks((long long)N * HC1, BS), BS, 0, stream>>>(OUT1, b1, N, HC1);
    k_edge_aggr<<<blks((long long)Ep * HH * ((C1 + 3) / 4), BS), BS, 0, stream>>>(
        P, Z, XW1, OUT1, src, dst, E0, Ep, C1);

    // ================= layer 2 =================
    k_cvt_f16<<<blks((long long)N * KP2, BS), BS, 0, stream>>>(OUT1, XH2, N, HC1, KP2, 1); // relu fused
    k_wT_f16 <<<blks((long long)NP2 * KP2, BS), BS, 0, stream>>>(W2, W2T, HC1, HC2, KP2, NP2);
    k_gemm_wmma<<<mtiles * (NP2 / 16), 32, 0, stream>>>(XH2, W2T, XW2, KP2, NP2 / 16, HC2, HC2);
    k_attn<<<blks((long long)N * HH, BS), BS, 0, stream>>>(XW2, a_s2, a_d2, ALS, ALD, N, C2);
    hipMemsetAsync(MENC, 0, (size_t)N * HH * 4, stream);
    hipMemsetAsync(Z,    0, (size_t)N * HH * 4, stream);
    k_edge_max<<<blks((long long)Ep * HH, BS), BS, 0, stream>>>(ALS, ALD, src, dst, MENC, E0, Ep);
    k_edge_p  <<<blks((long long)Ep * HH, BS), BS, 0, stream>>>(ALS, ALD, src, dst, MENC, P, Z, E0, Ep);
    k_init_bias<<<blks((long long)N * HC2, BS), BS, 0, stream>>>(OUT2, b2, N, HC2);
    k_edge_aggr<<<blks((long long)Ep * HH * ((C2 + 3) / 4), BS), BS, 0, stream>>>(
        P, Z, XW2, OUT2, src, dst, E0, Ep, C2);

    // ================= GraphNorm + pool + linear + sigmoid =================
    hipMemsetAsync(CNT,  0, GG * 4, stream);
    hipMemsetAsync(GSUM, 0, GG * HC2 * 4, stream);
    hipMemsetAsync(GVAR, 0, GG * HC2 * 4, stream);
    hipMemsetAsync(POOL, 0, GG * HC2 * 4, stream);
    k_counts   <<<blks(N, BS), BS, 0, stream>>>(batch, CNT, N);
    k_gn_sum   <<<blks((long long)N * HC2, BS), BS, 0, stream>>>(OUT2, batch, GSUM, N);
    k_gn_center<<<blks((long long)N * HC2, BS), BS, 0, stream>>>(OUT2, batch, GSUM, CNT, gn_ms, GVAR, N);
    k_gn_final <<<blks((long long)N * HC2, BS), BS, 0, stream>>>(OUT2, batch, GVAR, CNT, gn_w, gn_b, POOL, N);
    k_final    <<<blks(GG * 2, BS), BS, 0, stream>>>(POOL, CNT, lin_w, lin_b, (float*)d_out);
}
